// DGLMPN_30588757082627
// MI455X (gfx1250) — matrix operand order repaired and checked by reading, not compiled, not run on hardware
//
#include <hip/hip_runtime.h>
#include <hip/hip_bf16.h>

// ---------------------------------------------------------------------------
// D-MPNN (Chemprop-style) pipeline for MI455X (gfx1250, wave32, WMMA).
// All GEMMs run through v_wmma_f32_16x16x32_bf16 with fp32 accumulation.
// Segment sums use hardware fp32 global atomics (global_atomic_add_f32);
// messages stored as bf16 to halve HBM gather traffic (the bandwidth-bound
// part of this workload: ~2.5 GB over the 5 BP rounds @ 23.3 TB/s).
// ---------------------------------------------------------------------------

#define N_ATOMS   100000
#define N_EDGES   200000
#define N_LG      400000
#define N_GRAPHS  1024
#define ATOM_FDIM 39
#define BOND_FDIM 11
#define HIDDEN    300
#define DEPTH     6

#define HP   320   // HIDDEN padded to 20 tiles of 16 (and 10 K-steps of 32)
#define K1P  64    // (ATOM_FDIM+BOND_FDIM)=50 padded to 2 K-steps of 32
#define K3P  352   // (ATOM_FDIM+HIDDEN)=339 padded to 11 K-steps of 32

typedef __attribute__((ext_vector_type(16))) __bf16 v16bf;
typedef __attribute__((ext_vector_type(8)))  float  v8f;

__device__ __forceinline__ unsigned short f32_to_bf16(float f) {
  unsigned int u = __float_as_uint(f);
  u += 0x7fffu + ((u >> 16) & 1u);      // round to nearest even
  return (unsigned short)(u >> 16);
}
__device__ __forceinline__ float bf16_to_f32(unsigned short h) {
  return __uint_as_float(((unsigned int)h) << 16);
}

// Hardware float atomic add (global_atomic_add_f32), no CAS loop.
__device__ __forceinline__ void atomic_add_f32(float* p, float v) {
#if defined(__HIP_DEVICE_COMPILE__)
  unsafeAtomicAdd(p, v);
#else
  atomicAdd(p, v);
#endif
}

// ---- weight transpose+pad: Wt[n*KpadCols + k] = W[k*N + n], bf16 ----------
__global__ void transpose_pad_kernel(const float* __restrict__ W,
                                     unsigned short* __restrict__ Wt,
                                     int K, int N, int KpadCols, int NpadRows) {
  long i = (long)blockIdx.x * blockDim.x + threadIdx.x;
  long total = (long)NpadRows * KpadCols;
  if (i >= total) return;
  int n = (int)(i / KpadCols), k = (int)(i % KpadCols);
  float v = (k < K && n < N) ? W[(long)k * N + n] : 0.f;
  Wt[i] = f32_to_bf16(v);
}

// ---- line-graph node features: [atom[edge_src] | bond] -> bf16 [E x 64] ---
__global__ void build_feats_kernel(const float* __restrict__ atom,
                                   const float* __restrict__ bond,
                                   const int* __restrict__ esrc,
                                   unsigned short* __restrict__ feats) {
  long i = (long)blockIdx.x * blockDim.x + threadIdx.x;
  if (i >= (long)N_EDGES * K1P) return;
  int e = (int)(i >> 6), j = (int)(i & 63);
  float v = 0.f;
  if (j < ATOM_FDIM)                      v = atom[(long)esrc[e] * ATOM_FDIM + j];
  else if (j < ATOM_FDIM + BOND_FDIM)     v = bond[(long)e * BOND_FDIM + (j - ATOM_FDIM)];
  feats[i] = f32_to_bf16(v);
}

// ---- generic WMMA GEMM: C[M x HP] = act(A[M x Kpad] @ Bt^T + resid + bias)
// A is bf16 (or fp32, converted while staging into LDS). Bt is the weight
// matrix stored transposed+padded: Bt[n*ldb + k]. 128 threads = 4 waves;
// each block owns one 16-row M tile, waves stripe the 20 N tiles.
__global__ void __launch_bounds__(128)
gemm_wmma_kernel(const void* __restrict__ Av, int a_is_f32, int lda, int Kpad,
                 const unsigned short* __restrict__ Bt, int ldb,
                 const unsigned short* __restrict__ resid,  // [M x HP] bf16 or null
                 const float* __restrict__ bias, int bias_n, // per-col or null
                 float* __restrict__ outF,
                 unsigned short* __restrict__ outB,
                 unsigned short* __restrict__ outPre,       // pre-activation
                 int M, int relu) {
  extern __shared__ unsigned short ldsA[];   // 16 * Kpad bf16 elements
  const int row0 = blockIdx.x * 16;
  const int tid  = threadIdx.x;

  // Stage the 16 x Kpad A tile into LDS (fp32 -> bf16 convert if needed).
  const int total = 16 * Kpad;
  if (a_is_f32) {
    const float* A = (const float*)Av;
    for (int i = tid; i < total; i += 128) {
      int r = i / Kpad, c = i - r * Kpad;
      int gr = row0 + r;
      float v = (gr < M) ? A[(long)gr * lda + c] : 0.f;
      ldsA[i] = f32_to_bf16(v);
    }
  } else {
    const unsigned short* A = (const unsigned short*)Av;
    for (int i = tid; i < total; i += 128) {
      int r = i / Kpad, c = i - r * Kpad;
      int gr = row0 + r;
      ldsA[i] = (gr < M) ? A[(long)gr * lda + c] : (unsigned short)0;
    }
  }
  __syncthreads();

  const int wave = tid >> 5;
  const int lane = tid & 31;
  const int half = lane >> 4;   // K-half selector (A/B fragment layout)
  const int l15  = lane & 15;   // M row (A) / N column (B,C)

  for (int nt = wave; nt < HP / 16; nt += 4) {
    const int col0 = nt * 16;
    v8f acc = {0.f, 0.f, 0.f, 0.f, 0.f, 0.f, 0.f, 0.f};
    for (int kk = 0; kk < Kpad; kk += 32) {
      // A fragment (16x32 bf16): lane holds K = kk+half*8..+7 and kk+16+half*8..+7
      union { uint4 q[2]; v16bf v; } fa, fb;
      const unsigned short* ar = &ldsA[l15 * Kpad + kk + half * 8];
      fa.q[0] = *(const uint4*)(ar);
      fa.q[1] = *(const uint4*)(ar + 16);
      // B fragment (32x16 bf16): lane holds column col0+l15, K = kk+half*8..+7 and kk+16+half*8..+7
      const unsigned short* br = &Bt[(long)(col0 + l15) * ldb + kk + half * 8];
      fb.q[0] = *(const uint4*)(br);
      fb.q[1] = *(const uint4*)(br + 16);
      acc = __builtin_amdgcn_wmma_f32_16x16x32_bf16(
          false, fa.v, false, fb.v, (short)0, acc, false, false);
    }
    // Epilogue. C layout: element r -> row = r + 8*half, col = l15.
    for (int r = 0; r < 8; ++r) {
      int gr = row0 + r + half * 8;
      if (gr >= M) continue;
      int gc = col0 + l15;
      float v = acc[r];
      if (resid) v += bf16_to_f32(resid[(long)gr * HP + gc]);
      if (bias && gc < bias_n) v += bias[gc];
      if (outPre) outPre[(long)gr * HP + gc] = f32_to_bf16(v);
      if (relu && v < 0.f) v = 0.f;
      if (outF) outF[(long)gr * HP + gc] = v;
      if (outB) outB[(long)gr * HP + gc] = f32_to_bf16(v);
    }
  }
}

// ---- segment-sum over line graph: accum[lg_dst] += msg[lg_src] ------------
__global__ void scatter_lg_kernel(const unsigned short* __restrict__ msg,
                                  const int* __restrict__ src,
                                  const int* __restrict__ dst,
                                  float* __restrict__ accum, int nEdges) {
  long t = (long)blockIdx.x * blockDim.x + threadIdx.x;
  int e = (int)(t >> 6), sub = (int)(t & 63);
  if (e >= nEdges) return;
  const unsigned short* ms = msg + (long)src[e] * HP;
  float* ac = accum + (long)dst[e] * HP;
  for (int j = sub; j < HIDDEN; j += 64)
    atomic_add_f32(&ac[j], bf16_to_f32(ms[j]));
}

// ---- edge -> atom aggregation: m[edge_dst[e]] += msg[e] (identity source) -
__global__ void scatter_edge_kernel(const unsigned short* __restrict__ msg,
                                    const int* __restrict__ dst,
                                    float* __restrict__ m) {
  long t = (long)blockIdx.x * blockDim.x + threadIdx.x;
  int e = (int)(t >> 6), sub = (int)(t & 63);
  if (e >= N_EDGES) return;
  const unsigned short* ms = msg + (long)e * HP;
  float* ac = m + (long)dst[e] * HP;
  for (int j = sub; j < HIDDEN; j += 64)
    atomic_add_f32(&ac[j], bf16_to_f32(ms[j]));
}

// ---- readout input: [atom | m] -> bf16 [N_ATOMS x 352] ---------------------
__global__ void build_a3_kernel(const float* __restrict__ atom,
                                const float* __restrict__ m,
                                unsigned short* __restrict__ a3) {
  long i = (long)blockIdx.x * blockDim.x + threadIdx.x;
  if (i >= (long)N_ATOMS * K3P) return;
  int a = (int)(i / K3P), j = (int)(i % K3P);
  float v = 0.f;
  if (j < ATOM_FDIM)                  v = atom[(long)a * ATOM_FDIM + j];
  else if (j < ATOM_FDIM + HIDDEN)    v = m[(long)a * HP + (j - ATOM_FDIM)];
  a3[i] = f32_to_bf16(v);
}

// ---- per-graph pooling ----------------------------------------------------
__global__ void count_kernel(const int* __restrict__ gid, float* __restrict__ counts) {
  int a = blockIdx.x * blockDim.x + threadIdx.x;
  if (a < N_ATOMS) atomic_add_f32(&counts[gid[a]], 1.0f);
}

__global__ void pool_kernel(const float* __restrict__ h,
                            const int* __restrict__ gid,
                            float* __restrict__ out) {
  long t = (long)blockIdx.x * blockDim.x + threadIdx.x;
  if (t >= (long)N_ATOMS * HIDDEN) return;
  int a = (int)(t / HIDDEN), j = (int)(t % HIDDEN);
  atomic_add_f32(&out[(long)gid[a] * HIDDEN + j], h[(long)a * HP + j]);
}

__global__ void finalize_kernel(float* __restrict__ out, const float* __restrict__ counts) {
  int i = blockIdx.x * blockDim.x + threadIdx.x;
  if (i >= N_GRAPHS * HIDDEN) return;
  float c = counts[i / HIDDEN];
  out[i] = out[i] / (c > 1.0f ? c : 1.0f);
}

// ---------------------------------------------------------------------------
extern "C" void kernel_launch(void* const* d_in, const int* in_sizes, int n_in,
                              void* d_out, int out_size, void* d_ws, size_t ws_size,
                              hipStream_t stream) {
  const float* atom_features = (const float*)d_in[0];
  const float* bond_features = (const float*)d_in[1];
  const float* W_i = (const float*)d_in[2];
  const float* W_h = (const float*)d_in[3];
  const float* W_o = (const float*)d_in[4];
  const float* b_o = (const float*)d_in[5];
  const int* edge_src  = (const int*)d_in[6];
  const int* edge_dst  = (const int*)d_in[7];
  const int* lg_src    = (const int*)d_in[8];
  const int* lg_dst    = (const int*)d_in[9];
  const int* graph_ids = (const int*)d_in[10];
  float* out = (float*)d_out;

  // ---- workspace carving (256B aligned) ----
  size_t off = 0;
  char* base = (char*)d_ws;
  auto carve = [&](size_t bytes) -> void* {
    void* p = base + off;
    off += (bytes + 255) & ~(size_t)255;
    return p;
  };
  unsigned short* Wi_t  = (unsigned short*)carve((size_t)HP  * K1P * 2);
  unsigned short* Wh_t  = (unsigned short*)carve((size_t)HP  * HP  * 2);
  unsigned short* Wo_t  = (unsigned short*)carve((size_t)HP  * K3P * 2);
  unsigned short* feats = (unsigned short*)carve((size_t)N_EDGES * K1P * 2);
  unsigned short* msgin = (unsigned short*)carve((size_t)N_EDGES * HP * 2);  // pre-ReLU
  unsigned short* msg   = (unsigned short*)carve((size_t)N_EDGES * HP * 2);
  float*          accum = (float*)         carve((size_t)N_EDGES * HP * 4);
  float*          counts= (float*)         carve((size_t)N_GRAPHS * 4);
  // region reuse after the BP loop (lifetimes do not overlap):
  float*          mbuf  = accum;                          // [N_ATOMS x HP] fp32
  unsigned short* a3    = msgin;                          // [N_ATOMS x K3P] bf16
  float*          hbuf  = (float*)msg;                    // [N_ATOMS x HP] fp32

  // ---- init output / counts ----
  hipMemsetAsync(out, 0, (size_t)N_GRAPHS * HIDDEN * 4, stream);
  hipMemsetAsync(counts, 0, (size_t)N_GRAPHS * 4, stream);

  // ---- weight prep + input features ----
  {
    long t = (long)HP * K1P;
    transpose_pad_kernel<<<(unsigned)((t + 255) / 256), 256, 0, stream>>>(
        W_i, Wi_t, ATOM_FDIM + BOND_FDIM, HIDDEN, K1P, HP);
    t = (long)HP * HP;
    transpose_pad_kernel<<<(unsigned)((t + 255) / 256), 256, 0, stream>>>(
        W_h, Wh_t, HIDDEN, HIDDEN, HP, HP);
    t = (long)HP * K3P;
    transpose_pad_kernel<<<(unsigned)((t + 255) / 256), 256, 0, stream>>>(
        W_o, Wo_t, ATOM_FDIM + HIDDEN, HIDDEN, K3P, HP);
    t = (long)N_EDGES * K1P;
    build_feats_kernel<<<(unsigned)((t + 255) / 256), 256, 0, stream>>>(
        atom_features, bond_features, edge_src, feats);
  }

  const unsigned gemmBlocksE = (N_EDGES + 15) / 16;
  const unsigned gemmBlocksA = (N_ATOMS + 15) / 16;

  // ---- GEMM1: msg_input = feats @ W_i ; msg = relu(msg_input) ----
  gemm_wmma_kernel<<<gemmBlocksE, 128, 16 * K1P * 2, stream>>>(
      feats, 0, K1P, K1P, Wi_t, K1P,
      nullptr, nullptr, 0, nullptr, msg, msgin, N_EDGES, 1);

  // ---- loopy BP: DEPTH-1 rounds ----
  for (int step = 0; step < DEPTH - 1; ++step) {
    hipMemsetAsync(accum, 0, (size_t)N_EDGES * HP * 4, stream);
    {
      long t = (long)N_LG * 64;
      scatter_lg_kernel<<<(unsigned)((t + 255) / 256), 256, 0, stream>>>(
          msg, lg_src, lg_dst, accum, N_LG);
    }
    // msg = relu(msg_input + accum @ W_h); accum staged fp32 -> bf16 in LDS
    gemm_wmma_kernel<<<gemmBlocksE, 128, 16 * HP * 2, stream>>>(
        accum, 1, HP, HP, Wh_t, HP,
        msgin, nullptr, 0, nullptr, msg, nullptr, N_EDGES, 1);
  }

  // ---- edge -> atom aggregation: m[edge_dst[e]] += msg[e] ----
  hipMemsetAsync(mbuf, 0, (size_t)N_ATOMS * HP * 4, stream);
  {
    long t = (long)N_EDGES * 64;
    scatter_edge_kernel<<<(unsigned)((t + 255) / 256), 256, 0, stream>>>(
        msg, edge_dst, mbuf);
  }

  // ---- readout: h = relu([atom | m] @ W_o + b_o) ----
  {
    long t = (long)N_ATOMS * K3P;
    build_a3_kernel<<<(unsigned)((t + 255) / 256), 256, 0, stream>>>(
        atom_features, mbuf, a3);
  }
  gemm_wmma_kernel<<<gemmBlocksA, 128, 16 * K3P * 2, stream>>>(
      a3, 0, K3P, K3P, Wo_t, K3P,
      nullptr, b_o, HIDDEN, hbuf, nullptr, nullptr, N_ATOMS, 1);

  // ---- per-graph mean pooling ----
  count_kernel<<<(N_ATOMS + 255) / 256, 256, 0, stream>>>(graph_ids, counts);
  {
    long t = (long)N_ATOMS * HIDDEN;
    pool_kernel<<<(unsigned)((t + 255) / 256), 256, 0, stream>>>(hbuf, graph_ids, out);
  }
  finalize_kernel<<<(N_GRAPHS * HIDDEN + 255) / 256, 256, 0, stream>>>(out, counts);
}
